// SelfAttentionWithRelative_76991583748162
// MI455X (gfx1250) — compile-verified, matrix operand order checked by reading, not run on hardware
//
#include <hip/hip_runtime.h>
#include <hip/hip_bf16.h>
#include <stdint.h>

// ---------------------------------------------------------------------------
// MI455X (gfx1250) self-attention with relative position bias.
// All GEMMs via v_wmma_f32_16x16x32_bf16 (wave32), f32 accumulate.
// Attention stages K/V tiles into LDS with GLOBAL_LOAD_ASYNC_TO_LDS_B128
// (ASYNCcnt) shared by all 8 waves of the block.
// ---------------------------------------------------------------------------

#define B_SZ   32
#define SEQ    577
#define E      768
#define NH     12
#define HD     64
#define SPAD   640                 // seq padded: q-tiles of 128, k-tiles of 32
#define MROWS  (B_SZ * SEQ)        // 18464
#define MPAD   18560               // 145 * 128
#define KTILES 19                  // ceil(577/32)

typedef __attribute__((ext_vector_type(16))) __bf16  v16bf;
typedef __attribute__((ext_vector_type(8)))  float   v8f;
typedef __attribute__((ext_vector_type(4)))  unsigned u32x4;
typedef __attribute__((ext_vector_type(4)))  int      v4i;

__device__ __forceinline__ v8f wmma_bf16(v16bf a, v16bf b, v8f c) {
  return __builtin_amdgcn_wmma_f32_16x16x32_bf16(
      /*neg_a=*/false, a, /*neg_b=*/false, b,
      /*c_mod=*/(short)0, c, /*reuse_a=*/false, /*reuse_b=*/false);
}

__device__ __forceinline__ unsigned short f2bf(float f) {
  unsigned u = __float_as_uint(f);
  u += 0x7FFFu + ((u >> 16) & 1u);          // round-to-nearest-even
  return (unsigned short)(u >> 16);
}

// Load a 16x32 A/B fragment: two 16B chunks per lane (ISA 7.12.2 16-bit map:
// elems 0..7 <- K = half*8 + e ; elems 8..15 <- K = 16 + half*8 + (e-8)).
__device__ __forceinline__ v16bf load_frag(const unsigned short* c0,
                                           const unsigned short* c1) {
  union { v16bf v; u32x4 q[2]; } f;
  f.q[0] = *(const u32x4*)c0;
  f.q[1] = *(const u32x4*)c1;
  return f.v;
}

// ---- async global->LDS 16B copy (ASYNCcnt path), with portable fallback ----
#define HAS_ASYNC_LDS __has_builtin(__builtin_amdgcn_global_load_async_to_lds_b128)

__device__ __forceinline__ void cp16_to_lds(unsigned short* l, const unsigned short* g) {
#if HAS_ASYNC_LDS
  __builtin_amdgcn_global_load_async_to_lds_b128(
      (__attribute__((address_space(1))) v4i*)g,
      (__attribute__((address_space(3))) v4i*)l, 0, 0);
#else
  *(u32x4*)l = *(const u32x4*)g;   // sync fallback: b128 load + ds_store_b128
#endif
}

__device__ __forceinline__ void wait_async_lds() {
#if HAS_ASYNC_LDS
#if __has_builtin(__builtin_amdgcn_s_wait_asynccnt)
  __builtin_amdgcn_s_wait_asynccnt(0);
#else
  asm volatile("s_wait_asynccnt 0" ::: "memory");
#endif
#endif
}

// ---------------------------------------------------------------------------
// x (f32, [MROWS,E]) -> xb (bf16, [MPAD,E]), pad rows zeroed.
__global__ void k_cvt_x(const float* __restrict__ x, unsigned* __restrict__ xb) {
  int p = blockIdx.x * 256 + threadIdx.x;          // packed-pair index
  if (p >= MPAD * (E / 2)) return;
  int row = p / (E / 2);
  int c2  = (p - row * (E / 2)) * 2;
  unsigned v = 0;
  if (row < MROWS) {
    v = (unsigned)f2bf(x[(size_t)row * E + c2]) |
        ((unsigned)f2bf(x[(size_t)row * E + c2 + 1]) << 16);
  }
  xb[p] = v;
}

// W[k][n] f32 -> Wt[n][k] bf16, four matrices (q,k,v,o) back to back.
__global__ void k_cvt_w(const float* __restrict__ w0, const float* __restrict__ w1,
                        const float* __restrict__ w2, const float* __restrict__ w3,
                        unsigned* __restrict__ wt) {
  int t = blockIdx.x * 256 + threadIdx.x;          // 0 .. E*(E/2)
  int z = blockIdx.y;
  const float* w = (z == 0) ? w0 : (z == 1) ? w1 : (z == 2) ? w2 : w3;
  int n  = t / (E / 2);
  int kp = (t - n * (E / 2)) * 2;
  unsigned v = (unsigned)f2bf(w[(size_t)kp * E + n]) |
               ((unsigned)f2bf(w[(size_t)(kp + 1) * E + n]) << 16);
  wt[(size_t)z * (E * E / 2) + (size_t)n * (E / 2) + kp / 2] = v;
}

// ---------------------------------------------------------------------------
// QKV projection: each wave computes one 16x16 tile of xb @ W^T-rows.
// blockIdx.z selects Q/K/V; Q scaled by 1/sqrt(64); V stored transposed.
__global__ void __launch_bounds__(256)
k_gemm_qkv(const unsigned short* __restrict__ xb,
           const unsigned short* __restrict__ wt,
           unsigned short* __restrict__ qb,
           unsigned short* __restrict__ kb,
           unsigned short* __restrict__ vtr) {
  int wave = threadIdx.x >> 5, lane = threadIdx.x & 31;
  int half = lane >> 4, l16 = lane & 15;
  int m0 = blockIdx.x * 128 + wave * 16;
  int n0 = blockIdx.y * 16;
  int mode = blockIdx.z;
  const unsigned short* w = wt + (size_t)mode * E * E;

  v8f acc = {};
  const unsigned short* arow = xb + (size_t)(m0 + l16) * E + half * 8;
  const unsigned short* brow = w  + (size_t)(n0 + l16) * E + half * 8;
  #pragma unroll 4
  for (int k0 = 0; k0 < E; k0 += 32) {
    __builtin_prefetch(arow + k0 + 256, 0, 3);   // speculative; OOB dropped
    v16bf a = load_frag(arow + k0, arow + k0 + 16);
    v16bf b = load_frag(brow + k0, brow + k0 + 16);
    acc = wmma_bf16(a, b, acc);
  }

  int col = n0 + l16;
  int h = col >> 6, dc = col & 63;
  #pragma unroll
  for (int r = 0; r < 8; ++r) {
    int row = m0 + r + 8 * half;                   // C layout: M = r + 8*half
    if (row < MROWS) {
      int bidx = row / SEQ;
      int s = row - bidx * SEQ;
      float v = acc[r];
      if (mode == 0)
        qb[(((size_t)bidx * NH + h) * SPAD + s) * HD + dc] = f2bf(v * 0.125f);
      else if (mode == 1)
        kb[(((size_t)bidx * NH + h) * SPAD + s) * HD + dc] = f2bf(v);
      else
        vtr[(((size_t)bidx * NH + h) * HD + dc) * SPAD + s] = f2bf(v);
    }
  }
}

// ---------------------------------------------------------------------------
// Relative bias index (matches reference: 9=cls-cls, 10=cls-patch, 11=patch-cls)
__device__ __forceinline__ float relbias(const float* sb, bool qcls, int qy, int qx,
                                         int k, int ky, int kx) {
  if (qcls) return sb[(k == 0) ? 9 : 10];
  if (k == 0) return sb[11];
  int ry = qy - ky; ry = (ry < -1) ? -1 : ((ry > 1) ? 1 : ry);
  int rx = qx - kx; rx = (rx < -1) ? -1 : ((rx > 1) ? 1 : rx);
  return sb[(ry + 1) * 3 + (rx + 1)];
}

// Fused flash-attention: one wave owns 16 q-rows x d=64; the block cooperatively
// stages each 32-wide K/V tile into LDS via async loads, shared by all 8 waves.
__global__ void __launch_bounds__(256)
k_attn(const unsigned short* __restrict__ qb,
       const unsigned short* __restrict__ kb,
       const unsigned short* __restrict__ vtr,
       const float* __restrict__ bias_table,
       unsigned short* __restrict__ ao) {
  __shared__ float sb[12];
  __shared__ __align__(16) unsigned short sK[32 * 72];     // 32 k-rows x 64 d, stride 72
  __shared__ __align__(16) unsigned short sV[64 * 40];     // 64 d-rows x 32 k, stride 40
  __shared__ __align__(16) unsigned short sP[8][16 * 40];  // per-wave P tile, stride 40

  int wave = threadIdx.x >> 5, lane = threadIdx.x & 31;
  int half = lane >> 4, l16 = lane & 15;
  if (threadIdx.x < 12) sb[threadIdx.x] = bias_table[threadIdx.x];

  int b = blockIdx.z, h = blockIdx.y;
  int bh = b * NH + h;
  int qw = blockIdx.x * 128 + wave * 16;

  const unsigned short* qbase = qb  + (size_t)bh * SPAD * HD;
  const unsigned short* kbase = kb  + (size_t)bh * SPAD * HD;
  const unsigned short* vbase = vtr + (size_t)bh * HD * SPAD;

  // Q A-fragments (d = 0..31 and 32..63), resident for the whole k loop.
  const unsigned short* qrow = qbase + (size_t)(qw + l16) * HD + half * 8;
  v16bf aQ0 = load_frag(qrow,      qrow + 16);
  v16bf aQ1 = load_frag(qrow + 32, qrow + 48);

  // Per-thread staging slots: K tile = 256 x 16B chunks, V tile = 256 x 16B.
  int tkr = threadIdx.x >> 3;              // 0..31  (k row)
  int tkc = (threadIdx.x & 7) * 8;         // 0..56  (ushort col in d)
  int tvr = threadIdx.x >> 2;              // 0..63  (d row)
  int tvc = (threadIdx.x & 3) * 8;         // 0..24  (ushort col in k)
  const unsigned short* kg0 = kbase + (size_t)tkr * HD + tkc;
  const unsigned short* vg0 = vbase + (size_t)tvr * SPAD + tvc;
  unsigned short* lK = sK + tkr * 72 + tkc;
  unsigned short* lV = sV + tvr * 40 + tvc;

  int qy[8], qx[8]; bool qc[8];
  #pragma unroll
  for (int r = 0; r < 8; ++r) {
    int q = qw + r + 8 * half;
    qc[r] = (q == 0);
    int qi = (q > 0) ? (q - 1) : 0;
    qy[r] = qi / 24; qx[r] = qi - qy[r] * 24;
  }

  float m_i[8], l_i[8];
  v8f o0 = {}, o1 = {}, o2 = {}, o3 = {};
  #pragma unroll
  for (int r = 0; r < 8; ++r) { m_i[r] = -3.0e38f; l_i[r] = 0.0f; }

  unsigned short* P = sP[wave];

  for (int kt = 0; kt < KTILES; ++kt) {
    int k0 = kt * 32;
    if (kt) __syncthreads();               // previous tile fully consumed
    cp16_to_lds(lK, kg0 + (size_t)k0 * HD);
    cp16_to_lds(lV, vg0 + k0);
    wait_async_lds();
    __syncthreads();                       // tiles visible to all waves

    // scores = Q @ K^T for 32 k-positions (two 16-wide C tiles), from LDS
    const unsigned short* kr0 = sK + l16 * 72 + half * 8;
    const unsigned short* kr1 = sK + (16 + l16) * 72 + half * 8;
    v8f sc0 = {}, sc1 = {};
    sc0 = wmma_bf16(aQ0, load_frag(kr0,      kr0 + 16), sc0);
    sc0 = wmma_bf16(aQ1, load_frag(kr0 + 32, kr0 + 48), sc0);
    sc1 = wmma_bf16(aQ0, load_frag(kr1,      kr1 + 16), sc1);
    sc1 = wmma_bf16(aQ1, load_frag(kr1 + 32, kr1 + 48), sc1);

    int kc0 = k0 + l16, kc1 = k0 + 16 + l16;
    int ki0 = (kc0 > 0) ? kc0 - 1 : 0, ki1 = (kc1 > 0) ? kc1 - 1 : 0;
    int ky0 = ki0 / 24, kx0 = ki0 - ky0 * 24;
    int ky1 = ki1 / 24, kx1 = ki1 - ky1 * 24;
    bool v0 = (kc0 < SEQ), v1 = (kc1 < SEQ);

    #pragma unroll
    for (int r = 0; r < 8; ++r) {
      float s0 = sc0[r] + relbias(sb, qc[r], qy[r], qx[r], kc0, ky0, kx0);
      float s1 = sc1[r] + relbias(sb, qc[r], qy[r], qx[r], kc1, ky1, kx1);
      s0 = v0 ? s0 : -3.0e38f;
      s1 = v1 ? s1 : -3.0e38f;
      // row max across the 16 lanes of this half-wave (C layout: cols in lanes)
      float rm = fmaxf(s0, s1);
      rm = fmaxf(rm, __shfl_xor(rm, 1, 32));
      rm = fmaxf(rm, __shfl_xor(rm, 2, 32));
      rm = fmaxf(rm, __shfl_xor(rm, 4, 32));
      rm = fmaxf(rm, __shfl_xor(rm, 8, 32));
      float mn = fmaxf(m_i[r], rm);
      float sc = __expf(m_i[r] - mn);
      float p0 = __expf(s0 - mn);
      float p1 = __expf(s1 - mn);
      float rs = p0 + p1;
      rs += __shfl_xor(rs, 1, 32);
      rs += __shfl_xor(rs, 2, 32);
      rs += __shfl_xor(rs, 4, 32);
      rs += __shfl_xor(rs, 8, 32);
      l_i[r] = l_i[r] * sc + rs;
      m_i[r] = mn;
      o0[r] *= sc; o1[r] *= sc; o2[r] *= sc; o3[r] *= sc;
      int prow = r + 8 * half;                    // C-layout -> LDS row-major P
      P[prow * 40 + l16]      = f2bf(p0);
      P[prow * 40 + 16 + l16] = f2bf(p1);
    }

    // re-read P as an A fragment (16 q-rows x 32 k) from wave-private LDS
    const unsigned short* pr = P + l16 * 40 + half * 8;
    v16bf aP = load_frag(pr, pr + 16);

    // out += P @ V : B operand = rows of V^T tile in LDS (contiguous in k)
    const unsigned short* vr = sV + l16 * 40 + half * 8;
    o0 = wmma_bf16(aP, load_frag(vr,           vr + 16),           o0);
    o1 = wmma_bf16(aP, load_frag(vr + 16 * 40, vr + 16 * 40 + 16), o1);
    o2 = wmma_bf16(aP, load_frag(vr + 32 * 40, vr + 32 * 40 + 16), o2);
    o3 = wmma_bf16(aP, load_frag(vr + 48 * 40, vr + 48 * 40 + 16), o3);
  }

  #pragma unroll
  for (int r = 0; r < 8; ++r) {
    int q = qw + r + 8 * half;
    if (q < SEQ) {
      float inv = 1.0f / l_i[r];
      size_t ro = (size_t)(b * SEQ + q) * E + h * HD;
      ao[ro +  0 + l16] = f2bf(o0[r] * inv);
      ao[ro + 16 + l16] = f2bf(o1[r] * inv);
      ao[ro + 32 + l16] = f2bf(o2[r] * inv);
      ao[ro + 48 + l16] = f2bf(o3[r] * inv);
    }
  }
}

// ---------------------------------------------------------------------------
// Output projection: out = ao @ Wo + bo (f32 result)
__global__ void __launch_bounds__(256)
k_gemm_out(const unsigned short* __restrict__ ao,
           const unsigned short* __restrict__ wto,
           const float* __restrict__ bo,
           float* __restrict__ out) {
  int wave = threadIdx.x >> 5, lane = threadIdx.x & 31;
  int half = lane >> 4, l16 = lane & 15;
  int m0 = blockIdx.x * 128 + wave * 16;
  int n0 = blockIdx.y * 16;

  v8f acc = {};
  const unsigned short* arow = ao  + (size_t)(m0 + l16) * E + half * 8;
  const unsigned short* brow = wto + (size_t)(n0 + l16) * E + half * 8;
  #pragma unroll 4
  for (int k0 = 0; k0 < E; k0 += 32) {
    __builtin_prefetch(arow + k0 + 256, 0, 3);   // speculative; OOB dropped
    v16bf a = load_frag(arow + k0, arow + k0 + 16);
    v16bf b = load_frag(brow + k0, brow + k0 + 16);
    acc = wmma_bf16(a, b, acc);
  }

  int col = n0 + l16;
  float bias = bo[col];
  #pragma unroll
  for (int r = 0; r < 8; ++r) {
    int row = m0 + r + 8 * half;
    if (row < MROWS) out[(size_t)row * E + col] = acc[r] + bias;
  }
}

// ---------------------------------------------------------------------------
extern "C" void kernel_launch(void* const* d_in, const int* in_sizes, int n_in,
                              void* d_out, int out_size, void* d_ws, size_t ws_size,
                              hipStream_t stream) {
  const float* x  = (const float*)d_in[0];
  const float* Wq = (const float*)d_in[1];
  const float* Wk = (const float*)d_in[2];
  const float* Wv = (const float*)d_in[3];
  const float* Wo = (const float*)d_in[4];
  const float* bo = (const float*)d_in[5];
  const float* bt = (const float*)d_in[6];
  float* out = (float*)d_out;

  char* ws = (char*)d_ws;
  size_t off = 0;
  unsigned short* xb  = (unsigned short*)(ws + off); off += (size_t)MPAD * E * 2;
  unsigned short* wt  = (unsigned short*)(ws + off); off += (size_t)4 * E * E * 2;
  unsigned short* qb  = (unsigned short*)(ws + off); off += (size_t)B_SZ * NH * SPAD * HD * 2;
  unsigned short* kb  = (unsigned short*)(ws + off); off += (size_t)B_SZ * NH * SPAD * HD * 2;
  unsigned short* vtr = (unsigned short*)(ws + off); off += (size_t)B_SZ * NH * HD * SPAD * 2;
  unsigned short* ao  = (unsigned short*)(ws + off); off += (size_t)MPAD * E * 2;

  (void)hipMemsetAsync(d_ws, 0, off, stream);

  k_cvt_x<<<(MPAD * (E / 2) + 255) / 256, 256, 0, stream>>>(x, (unsigned*)xb);
  k_cvt_w<<<dim3(E * (E / 2) / 256, 4), 256, 0, stream>>>(Wq, Wk, Wv, Wo, (unsigned*)wt);
  k_gemm_qkv<<<dim3(MPAD / 128, E / 16, 3), 256, 0, stream>>>(xb, wt, qb, kb, vtr);
  k_attn<<<dim3(5, NH, B_SZ), 256, 0, stream>>>(qb, kb, vtr, bt, ao);
  k_gemm_out<<<dim3(MPAD / 128, E / 16), 256, 0, stream>>>(ao, wt + (size_t)3 * E * E, bo, out);
}
